// GraphTransformerNet_40295383171718
// MI455X (gfx1250) — matrix operand order repaired and checked by reading, not compiled, not run on hardware
//
#include <hip/hip_runtime.h>
#include <hip/hip_bf16.h>

// ---------------------------------------------------------------------------
// GraphTransformerNet forward for MI455X (gfx1250), wave32 + WMMA bf16.
//
// Roofline: ~183 GFLOP of GEMM vs HBM traffic dominated by E=256k x 128/256
// edge tensors -> memory bound at 23.3 TB/s. Strategy:
//   * all large activations stored bf16 (halves the dominant traffic term)
//   * bf16 WMMA GEMMs (v_wmma_f32_16x16x32_bf16), f32 accumulate
//   * epilogues (bias / ReLU / GELU / residual+BN-eval) fused into the store
//   * atomic scatter kernels for the per-destination edge softmax
// ---------------------------------------------------------------------------

typedef __attribute__((ext_vector_type(16))) __bf16 bf16x16;
typedef __attribute__((ext_vector_type(8)))  __bf16 bf16x8;
typedef __attribute__((ext_vector_type(8)))  float  f32x8;

#define EPI_NONE  0
#define EPI_RELU  1
#define EPI_BNRES 2
#define EPI_GELU  3

// monotone encoding of float into unsigned for atomicMax-based segment max
__device__ __forceinline__ unsigned fenc(float f) {
  unsigned u = __float_as_uint(f);
  return (u & 0x80000000u) ? ~u : (u | 0x80000000u);
}
__device__ __forceinline__ float fdec(unsigned u) {
  unsigned v = (u & 0x80000000u) ? (u & 0x7FFFFFFFu) : ~u;
  return __uint_as_float(v);
}

// A-fragment loaders (CDNA5 ISA 7.12.2 16-bit A layout):
//   lane L row=(L&15); halves 0..7 = K base+0..7 ; halves 8..15 = K base+16..23
// bf16 source: two aligned 16B vector loads, no conversion.
__device__ __forceinline__ bf16x16 load_afrag(const __bf16* ap) {
  bf16x8 lo = *(const bf16x8*)ap;
  bf16x8 hi = *(const bf16x8*)(ap + 16);
  return __builtin_shufflevector(lo, hi, 0, 1, 2, 3, 4, 5, 6, 7,
                                         8, 9, 10, 11, 12, 13, 14, 15);
}
// f32 source: hardware cvt per element (only used for the small wV tensor).
__device__ __forceinline__ bf16x16 load_afrag(const float* ap) {
  bf16x16 r;
#pragma unroll
  for (int i = 0; i < 8; ++i) r[i] = (__bf16)ap[i];
#pragma unroll
  for (int i = 0; i < 8; ++i) r[8 + i] = (__bf16)ap[16 + i];
  return r;
}

// ---------------------------------------------------------------------------
// Generic tiled WMMA GEMM:  Out[M,NC] = epilogue(A[M,K] @ W[K,NC])
// 256 threads = 8 waves. Block tile = 128 rows x NC cols; each wave owns a
// 16-row strip and NT=NC/16 column tiles. W (f32 in global) is staged
// transposed (bf16) into LDS in 64-deep K chunks (pad to kill bank conflicts).
// ---------------------------------------------------------------------------
template<typename AT, int K, int NC, int EPI>
__global__ __launch_bounds__(256)
void gemm_wmma(const AT* __restrict__ A, const float* __restrict__ W,
               const float* __restrict__ bias, const __bf16* __restrict__ res,
               const float* __restrict__ bng, const float* __restrict__ bnb,
               const float* __restrict__ bnm, const float* __restrict__ bnv,
               __bf16* __restrict__ Out, int M)
{
  extern __shared__ __bf16 ldsWT[];      // [NC][72] : W^T bf16 for one K-chunk
  const int KP = 72;                     // 64 + 8 pad
  const int NT = NC / 16;
  const int tid  = threadIdx.x;
  const int wave = tid >> 5;
  const int lane = tid & 31;
  const int lhi  = lane >> 4;            // 0 or 1
  const int llo  = lane & 15;
  const int rowbase = blockIdx.x * 128 + wave * 16;

  int arow = rowbase + llo;              // A row this lane feeds to WMMA
  if (arow >= M) arow = M - 1;           // clamp; bogus rows masked at store
  const AT* aptr = A + (size_t)arow * K;

  f32x8 acc[NT];
#pragma unroll
  for (int t = 0; t < NT; ++t) acc[t] = (f32x8)0.0f;

  for (int kc = 0; kc < K; kc += 64) {
    __syncthreads();
    for (int idx = tid; idx < 64 * NC; idx += 256) {
      int kl = idx / NC, c = idx % NC;
      ldsWT[c * KP + kl] = (__bf16)W[(size_t)(kc + kl) * NC + c];
    }
    __syncthreads();
#pragma unroll
    for (int k0 = 0; k0 < 64; k0 += 32) {
      const AT* ap = aptr + kc + k0 + lhi * 8;
      __builtin_prefetch(ap + 64, 0, 1);         // global_prefetch next chunk
      bf16x16 af = load_afrag(ap);
#pragma unroll
      for (int t = 0; t < NT; ++t) {
        // B (32x16): lane L col=(L&15); halves 0..7 = K k0+(L>>4)*8+0..7 ; halves 8..15 = K k0+16+(L>>4)*8+0..7
        const __bf16* bp = ldsWT + (t * 16 + llo) * KP + k0 + lhi * 8;
        bf16x16 bfr;
#pragma unroll
        for (int i = 0; i < 16; ++i) bfr[i] = bp[(i & 7) + ((i >> 3) << 4)];
        acc[t] = __builtin_amdgcn_wmma_f32_16x16x32_bf16(
            false, af, false, bfr, (short)0, acc[t], false, false);
      }
    }
  }

  // fused epilogue; lane stores rows rowbase + lhi*8 + r at col t*16 + llo
#pragma unroll
  for (int t = 0; t < NT; ++t) {
    const int c = t * 16 + llo;
    float bi = 0.0f;
    if (EPI != EPI_NONE) bi = bias[c];
    float scale = 1.0f, shift = 0.0f;
    if (EPI == EPI_BNRES) {
      scale = bng[c] * rsqrtf(bnv[c] + 1e-5f);
      shift = bnb[c] - bnm[c] * scale;
    }
#pragma unroll
    for (int r = 0; r < 8; ++r) {
      int row = rowbase + lhi * 8 + r;
      if (row < M) {
        float v = acc[t][r] + bi;
        if (EPI == EPI_RELU)  v = fmaxf(v, 0.0f);
        if (EPI == EPI_GELU)  v = 0.5f * v * (1.0f + erff(v * 0.70710678118f));
        if (EPI == EPI_BNRES) v = ((float)res[(size_t)row * NC + c] + v) * scale + shift;
        Out[(size_t)row * NC + c] = (__bf16)v;
      }
    }
  }
}

// ---------------------------------------------------------------------------
// Elementwise / scatter kernels
// ---------------------------------------------------------------------------
__global__ __launch_bounds__(256)
void k_embed_h(const int* __restrict__ nf, const float* __restrict__ emb,
               const float* __restrict__ lap, const float* __restrict__ lw,
               const float* __restrict__ lb, __bf16* __restrict__ h, int cnt)
{
  int idx = blockIdx.x * 256 + threadIdx.x;
  if (idx >= cnt) return;
  int n = idx >> 7, c = idx & 127;
  float acc = emb[nf[n] * 128 + c] + lb[c];
  const float* lp = lap + (size_t)n * 8;
#pragma unroll
  for (int p = 0; p < 8; ++p) acc += lp[p] * lw[p * 128 + c];
  h[idx] = (__bf16)acc;
}

__global__ __launch_bounds__(256)
void k_embed_e(const int* __restrict__ ef, const float* __restrict__ emb,
               __bf16* __restrict__ e, int cnt)
{
  int idx = blockIdx.x * 256 + threadIdx.x;
  if (idx >= cnt) return;
  int n = idx >> 7, c = idx & 127;
  e[idx] = (__bf16)emb[ef[n] * 128 + c];
}

// per-(edge,head) attention score added in-place onto proj_e (etmp, bf16)
__global__ __launch_bounds__(256)
void k_score(const __bf16* __restrict__ Q, const __bf16* __restrict__ Kb,
             const int* __restrict__ src, const int* __restrict__ dst,
             __bf16* __restrict__ etmp, int cnt)
{
  int idx = blockIdx.x * 256 + threadIdx.x;
  if (idx >= cnt) return;
  int e = idx >> 3, hd = idx & 7;
  int s = src[e], d = dst[e];
  const __bf16* kp = Kb + (size_t)s * 128 + hd * 16;
  const __bf16* qp = Q  + (size_t)d * 128 + hd * 16;
  float acc = 0.0f;
#pragma unroll
  for (int i = 0; i < 16; ++i) acc += (float)kp[i] * (float)qp[i];
  acc *= 0.25f;                              // 1/sqrt(DH)
  __bf16* ep = etmp + (size_t)e * 128 + hd * 16;
#pragma unroll
  for (int i = 0; i < 16; ++i) ep[i] = (__bf16)((float)ep[i] + acc);
}

__global__ __launch_bounds__(256)
void k_init(unsigned* __restrict__ mx, float* __restrict__ sm,
            float* __restrict__ wV, int cnt)
{
  int idx = blockIdx.x * 256 + threadIdx.x;
  if (idx >= cnt) return;
  mx[idx] = 0x007FFFFFu;                     // fenc(-inf)
  sm[idx] = 0.0f;
  wV[idx] = 0.0f;
}

__global__ __launch_bounds__(256)
void k_segmax(const __bf16* __restrict__ etmp, const int* __restrict__ dst,
              unsigned* __restrict__ mx, int cnt)
{
  int idx = blockIdx.x * 256 + threadIdx.x;
  if (idx >= cnt) return;
  int e = idx >> 7, c = idx & 127;
  atomicMax(mx + (size_t)dst[e] * 128 + c, fenc((float)etmp[idx]));
}

__global__ __launch_bounds__(256)
void k_segsum(const __bf16* __restrict__ etmp, const int* __restrict__ dst,
              const unsigned* __restrict__ mx, float* __restrict__ sm, int cnt)
{
  int idx = blockIdx.x * 256 + threadIdx.x;
  if (idx >= cnt) return;
  int e = idx >> 7, c = idx & 127;
  size_t dc = (size_t)dst[e] * 128 + c;
  atomicAdd(sm + dc, __expf((float)etmp[idx] - fdec(mx[dc])));
}

__global__ __launch_bounds__(256)
void k_segagg(const __bf16* __restrict__ etmp, const __bf16* __restrict__ V,
              const int* __restrict__ src, const int* __restrict__ dst,
              const unsigned* __restrict__ mx, const float* __restrict__ sm,
              float* __restrict__ wV, int cnt)
{
  int idx = blockIdx.x * 256 + threadIdx.x;
  if (idx >= cnt) return;
  int e = idx >> 7, c = idx & 127;
  size_t dc = (size_t)dst[e] * 128 + c;
  float alpha = __expf((float)etmp[idx] - fdec(mx[dc])) / sm[dc];
  atomicAdd(wV + dc, (float)V[(size_t)src[e] * 128 + c] * alpha);
}

__global__ __launch_bounds__(256)
void k_boxes(const __bf16* __restrict__ x2, const float* __restrict__ w3,
             const float* __restrict__ b3, float* __restrict__ out, int cnt)
{
  int idx = blockIdx.x * 256 + threadIdx.x;
  if (idx >= cnt) return;
  int n = idx >> 2, j = idx & 3;
  float acc = b3[j];
  const __bf16* xp = x2 + (size_t)n * 32;
#pragma unroll
  for (int i = 0; i < 32; ++i) acc += (float)xp[i] * w3[i * 4 + j];
  out[idx] = 1.0f / (1.0f + __expf(-acc));
}

// ---------------------------------------------------------------------------
// Host side
// ---------------------------------------------------------------------------
template<typename AT, int K, int NC, int EPI>
static void gemm(const AT* A, const float* W, const float* bias,
                 const __bf16* res, const float* g, const float* b,
                 const float* m, const float* v, __bf16* out, int M,
                 hipStream_t s)
{
  dim3 grid((unsigned)((M + 127) / 128)), block(256);
  size_t lds = (size_t)NC * 72 * sizeof(__bf16);
  hipLaunchKernelGGL(HIP_KERNEL_NAME(gemm_wmma<AT, K, NC, EPI>), grid, block, lds, s,
                     A, W, bias, res, g, b, m, v, out, M);
}

static inline dim3 ewgrid(long long cnt) { return dim3((unsigned)((cnt + 255) / 256)); }

extern "C" void kernel_launch(void* const* d_in, const int* in_sizes, int n_in,
                              void* d_out, int out_size, void* d_ws, size_t ws_size,
                              hipStream_t stream)
{
  (void)n_in; (void)out_size; (void)ws_size;
  // --- inputs (setup_inputs() dict insertion order, recursive) ---
  const int*   node_feat = (const int*)  d_in[0];
  const int*   edge_feat = (const int*)  d_in[1];
  const int*   src       = (const int*)  d_in[2];
  const int*   dst       = (const int*)  d_in[3];
  const float* lap       = (const float*)d_in[4];
  const float* emb_h     = (const float*)d_in[5];
  const float* emb_e     = (const float*)d_in[6];
  const float* lap_w     = (const float*)d_in[7];
  const float* lap_b     = (const float*)d_in[8];
  // layers (each with leading L=3 axis): 9..40
  const float* Wq   = (const float*)d_in[9];
  const float* Wk   = (const float*)d_in[10];
  const float* Wv   = (const float*)d_in[11];
  const float* We   = (const float*)d_in[12];
  const float* OhW  = (const float*)d_in[13];
  const float* OhB  = (const float*)d_in[14];
  const float* OeW  = (const float*)d_in[15];
  const float* OeB  = (const float*)d_in[16];
  const float* Fh1W = (const float*)d_in[17];
  const float* Fh1B = (const float*)d_in[18];
  const float* Fh2W = (const float*)d_in[19];
  const float* Fh2B = (const float*)d_in[20];
  const float* Fe1W = (const float*)d_in[21];
  const float* Fe1B = (const float*)d_in[22];
  const float* Fe2W = (const float*)d_in[23];
  const float* Fe2B = (const float*)d_in[24];
  const float* bn1h[4] = { (const float*)d_in[25], (const float*)d_in[26],
                           (const float*)d_in[27], (const float*)d_in[28] };
  const float* bn1e[4] = { (const float*)d_in[29], (const float*)d_in[30],
                           (const float*)d_in[31], (const float*)d_in[32] };
  const float* bn2h[4] = { (const float*)d_in[33], (const float*)d_in[34],
                           (const float*)d_in[35], (const float*)d_in[36] };
  const float* bn2e[4] = { (const float*)d_in[37], (const float*)d_in[38],
                           (const float*)d_in[39], (const float*)d_in[40] };
  const float* rw1 = (const float*)d_in[41];
  const float* rb1 = (const float*)d_in[42];
  const float* rw2 = (const float*)d_in[43];
  const float* rb2 = (const float*)d_in[44];
  const float* rw3 = (const float*)d_in[45];
  const float* rb3 = (const float*)d_in[46];

  const int N_ = in_sizes[0];          // 40000
  const int E_ = in_sizes[1];          // 256000
  const long long NH = (long long)N_ * 128;
  const long long EH = (long long)E_ * 128;
  const int CHUNK = 65536;             // FFN-e row chunk (bounds zbuf)

  // --- workspace carve (~270 MB), 256B aligned slices ---
  char* wsb = (char*)d_ws;
  size_t off = 0;
  auto carve = [&](size_t bytes) -> void* {
    void* r = wsb + off;
    off += (bytes + 255) & ~(size_t)255;
    return r;
  };
  __bf16*   h   = (__bf16*)carve(NH * 2);
  __bf16*   Qb  = (__bf16*)carve(NH * 2);
  __bf16*   Kb  = (__bf16*)carve(NH * 2);
  __bf16*   Vb  = (__bf16*)carve(NH * 2);
  float*    wV  = (float*) carve(NH * 4);
  float*    sm  = (float*) carve(NH * 4);
  unsigned* mx  = (unsigned*)carve(NH * 4);
  __bf16*   e   = (__bf16*)carve(EH * 2);
  __bf16*   etm = (__bf16*)carve(EH * 2);      // proj_e -> score -> e_out
  __bf16*   zb  = (__bf16*)carve((size_t)CHUNK * 256 * 2); // FFN hidden
  __bf16*   x1  = zb;                          // regressor reuse
  __bf16*   x2  = zb + (size_t)N_ * 64;

  // --- embeddings ---
  hipLaunchKernelGGL(k_embed_h, ewgrid(NH), dim3(256), 0, stream,
                     node_feat, emb_h, lap, lap_w, lap_b, h, (int)NH);
  hipLaunchKernelGGL(k_embed_e, ewgrid(EH), dim3(256), 0, stream,
                     edge_feat, emb_e, e, (int)EH);

  // --- 3 GraphTransformer layers ---
  for (int l = 0; l < 3; ++l) {
    const size_t wOff = (size_t)l * 128 * 128;   // D*D weights
    const size_t bOff = (size_t)l * 128;         // D biases / bn
    const size_t fOff = (size_t)l * 128 * 256;   // D*2D weights
    const size_t f1bO = (size_t)l * 256;         // 2D bias

    // projections (no bias)
    gemm<__bf16,128,128,EPI_NONE>(h, Wq + wOff, nullptr, nullptr, 0,0,0,0, Qb, N_, stream);
    gemm<__bf16,128,128,EPI_NONE>(h, Wk + wOff, nullptr, nullptr, 0,0,0,0, Kb, N_, stream);
    gemm<__bf16,128,128,EPI_NONE>(h, Wv + wOff, nullptr, nullptr, 0,0,0,0, Vb, N_, stream);
    gemm<__bf16,128,128,EPI_NONE>(e, We + wOff, nullptr, nullptr, 0,0,0,0, etm, E_, stream);

    // score = K[src].Q[dst]/4 broadcast-added onto proj_e  (etm == e_out)
    hipLaunchKernelGGL(k_score, ewgrid((long long)E_ * 8), dim3(256), 0, stream,
                       Qb, Kb, src, dst, etm, E_ * 8);

    // edge softmax over incoming edges per (node, channel) + weighted V scatter
    hipLaunchKernelGGL(k_init,   ewgrid(NH), dim3(256), 0, stream, mx, sm, wV, (int)NH);
    hipLaunchKernelGGL(k_segmax, ewgrid(EH), dim3(256), 0, stream, etm, dst, mx, (int)EH);
    hipLaunchKernelGGL(k_segsum, ewgrid(EH), dim3(256), 0, stream, etm, dst, mx, sm, (int)EH);
    hipLaunchKernelGGL(k_segagg, ewgrid(EH), dim3(256), 0, stream,
                       etm, Vb, src, dst, mx, sm, wV, (int)EH);

    // O-projections with fused residual + BN(eval)
    gemm<float ,128,128,EPI_BNRES>(wV,  OhW + wOff, OhB + bOff, h,
                            bn1h[0] + bOff, bn1h[1] + bOff, bn1h[2] + bOff, bn1h[3] + bOff,
                            h, N_, stream);
    gemm<__bf16,128,128,EPI_BNRES>(etm, OeW + wOff, OeB + bOff, e,
                            bn1e[0] + bOff, bn1e[1] + bOff, bn1e[2] + bOff, bn1e[3] + bOff,
                            e, E_, stream);

    // FFN on h
    gemm<__bf16,128,256,EPI_RELU >(h,  Fh1W + fOff, Fh1B + f1bO, nullptr, 0,0,0,0, zb, N_, stream);
    gemm<__bf16,256,128,EPI_BNRES>(zb, Fh2W + fOff, Fh2B + bOff, h,
                            bn2h[0] + bOff, bn2h[1] + bOff, bn2h[2] + bOff, bn2h[3] + bOff,
                            h, N_, stream);

    // FFN on e, chunked to bound scratch
    for (int base = 0; base < E_; base += CHUNK) {
      int cm = (E_ - base < CHUNK) ? (E_ - base) : CHUNK;
      const __bf16* ec = e + (size_t)base * 128;
      __bf16*       eo = e + (size_t)base * 128;
      gemm<__bf16,128,256,EPI_RELU >(ec, Fe1W + fOff, Fe1B + f1bO, nullptr, 0,0,0,0, zb, cm, stream);
      gemm<__bf16,256,128,EPI_BNRES>(zb, Fe2W + fOff, Fe2B + bOff, ec,
                            bn2e[0] + bOff, bn2e[1] + bOff, bn2e[2] + bOff, bn2e[3] + bOff,
                            eo, cm, stream);
    }
  }

  // --- box regressor: GELU MLP 128->64->32, then sigmoid 32->4 ---
  gemm<__bf16,128,64,EPI_GELU>(h,  rw1, rb1, nullptr, 0,0,0,0, x1, N_, stream);
  gemm<__bf16, 64,32,EPI_GELU>(x1, rw2, rb2, nullptr, 0,0,0,0, x2, N_, stream);
  hipLaunchKernelGGL(k_boxes, ewgrid((long long)N_ * 4), dim3(256), 0, stream,
                     x2, rw3, rb3, (float*)d_out, N_ * 4);
}
